// SparseTransformer_12025908428867
// MI455X (gfx1250) — matrix operand order repaired and checked by reading, not compiled, not run on hardware
//
#include <hip/hip_runtime.h>
#include <hip/hip_bf16.h>
#include <hip/hip_fp16.h>

// ---------------------------------------------------------------------------
// Types for the CDNA5 WMMA path (wave32, 16x16x32 bf16 -> f32 accumulate)
// ---------------------------------------------------------------------------
typedef __attribute__((ext_vector_type(8)))  float  v8f;
typedef __attribute__((ext_vector_type(16))) __bf16 v16bf;
typedef __attribute__((ext_vector_type(8)))  __bf16 v8bf;

// TDM descriptor vector types (probe-verified builtin operand shapes)
typedef __attribute__((ext_vector_type(4))) unsigned int u32x4;
typedef __attribute__((ext_vector_type(8))) int          i32x8;
typedef __attribute__((ext_vector_type(4))) int          i32x4;

#if defined(__AMDGCN__) && __has_builtin(__builtin_amdgcn_tensor_load_to_lds)
#define HAVE_TDM 1
#else
#define HAVE_TDM 0
#endif

// Raw (unconverted) load of 8 consecutive K elements into float4 slots.
// fp32 source: 2x b128. bf16 source: 1x b128 (8 bf16 = 16 bytes).
template <typename T>
__device__ __forceinline__ void loadraw8(const T* __restrict__ p, float4* r) {
  if constexpr (sizeof(T) == 4) {
    r[0] = *(const float4*)p;
    r[1] = *(const float4*)(p + 4);
  } else {
    r[0] = *(const float4*)p;
  }
}

// Convert one raw 8-element slot pair to 8 bf16 values.
template <typename T>
__device__ __forceinline__ void cvt8(const float4* r, __bf16* d) {
  if constexpr (sizeof(T) == 4) {
    d[0] = (__bf16)r[0].x; d[1] = (__bf16)r[0].y;
    d[2] = (__bf16)r[0].z; d[3] = (__bf16)r[0].w;
    d[4] = (__bf16)r[1].x; d[5] = (__bf16)r[1].y;
    d[6] = (__bf16)r[1].z; d[7] = (__bf16)r[1].w;
  } else {
    v8bf v;
    __builtin_memcpy(&v, &r[0], 16);
#pragma unroll
    for (int i = 0; i < 8; ++i) d[i] = v[i];
  }
}

// ---------------------------------------------------------------------------
// C[M,N] = (A[M,K] * Bw[N,K]^T) * scale + bias[N]
// One wave computes a 16x64 tile; 2-stage software pipeline over K (step 64):
// all 10 b128 loads of a stage are issued in a batch (s_clause), the next
// stage's loads are in flight while the current stage converts and WMMAs.
// Requires: M % 16 == 0, N % 64 == 0, K % 64 == 0 (true for every call here).
// ---------------------------------------------------------------------------
template <typename TA, typename TB, typename TC>
__global__ __launch_bounds__(32)
void gemm_abt_wmma(const TA* __restrict__ A, const TB* __restrict__ Bw,
                   const float* __restrict__ bias,
                   const float* __restrict__ scale_ptr, int scale_is_rcp,
                   TC* __restrict__ C, int M, int N, int K) {
  const int lane = (int)(threadIdx.x & 31u);
  const int half = lane >> 4;
  const int l    = lane & 15;
  const int mbase = (int)blockIdx.y << 4;
  const int nbase = (int)blockIdx.x << 6;

  const TA* __restrict__ pa  = A  + (size_t)(mbase + l) * (size_t)K;
  const TB* __restrict__ pb0 = Bw + (size_t)(nbase +  0 + l) * (size_t)K;
  const TB* __restrict__ pb1 = Bw + (size_t)(nbase + 16 + l) * (size_t)K;
  const TB* __restrict__ pb2 = Bw + (size_t)(nbase + 32 + l) * (size_t)K;
  const TB* __restrict__ pb3 = Bw + (size_t)(nbase + 48 + l) * (size_t)K;

  const int ka0 = half * 8;        // first 8 K values this lane owns (A)
  const int ka1 = 16 + half * 8;   // second 8 K values this lane owns (A)
  const int kb0 = half * 8;        // B: same K order as A (kb0, kb0 + 16)

  v8f acc[4] = {};

  float4 ra[2][2][2];
  float4 rb[2][4][2][2];

  auto load_stage = [&](int s, int kk) {
    loadraw8<TA>(pa + kk + ka0, ra[s][0]);
    loadraw8<TA>(pa + kk + ka1, ra[s][1]);
    loadraw8<TB>(pb0 + kk + kb0,     rb[s][0][0]);
    loadraw8<TB>(pb0 + kk + kb0 + 16, rb[s][0][1]);
    loadraw8<TB>(pb1 + kk + kb0,     rb[s][1][0]);
    loadraw8<TB>(pb1 + kk + kb0 + 16, rb[s][1][1]);
    loadraw8<TB>(pb2 + kk + kb0,     rb[s][2][0]);
    loadraw8<TB>(pb2 + kk + kb0 + 16, rb[s][2][1]);
    loadraw8<TB>(pb3 + kk + kb0,     rb[s][3][0]);
    loadraw8<TB>(pb3 + kk + kb0 + 16, rb[s][3][1]);
  };

  auto compute_stage = [&](int s) {
    v16bf af;
    {
      __bf16 tt[16];
      cvt8<TA>(ra[s][0], tt);
      cvt8<TA>(ra[s][1], tt + 8);
#pragma unroll
      for (int i = 0; i < 16; ++i) af[i] = tt[i];
    }
#pragma unroll
    for (int t = 0; t < 4; ++t) {
      v16bf bfr;
      {
        __bf16 tt[16];
        cvt8<TB>(rb[s][t][0], tt);
        cvt8<TB>(rb[s][t][1], tt + 8);
#pragma unroll
        for (int i = 0; i < 16; ++i) bfr[i] = tt[i];
      }
      acc[t] = __builtin_amdgcn_wmma_f32_16x16x32_bf16(
          false, af, false, bfr, (short)0, acc[t], false, false);
    }
  };

  load_stage(0, 0);
  for (int kk = 0; kk < K; kk += 64) {
    load_stage(1, kk + 32);
    compute_stage(0);
    if (kk + 64 < K) load_stage(0, kk + 64);
    compute_stage(1);
  }

  float scale = 1.0f;
  if (scale_ptr) { scale = *scale_ptr; if (scale_is_rcp) scale = 1.0f / scale; }

#pragma unroll
  for (int t = 0; t < 4; ++t) {
    const int col = nbase + t * 16 + l;
    const float bv = bias ? bias[col] : 0.0f;
#pragma unroll
    for (int i = 0; i < 8; ++i) {
      const int row = mbase + half * 8 + i;
      C[(size_t)row * (size_t)N + col] = (TC)(acc[t][i] * scale + bv);
    }
  }
}

// ---------------------------------------------------------------------------
// Vt[c][r] = (bf16) V[r][c] — 32x32 tile transpose.
// TDM path: one wave issues a Tensor-Data-Mover descriptor that DMAs the
// 32x32 fp32 tile global->LDS with hardware padding (pad every 32 DWORDs by
// 1 DWORD => LDS row stride 33 floats, conflict-free column reads), waits on
// TENSORcnt, then the whole block stores the transposed tile as bf16.
// ---------------------------------------------------------------------------
__global__ __launch_bounds__(256)
void transpose_to_bf16(const float* __restrict__ V, __bf16* __restrict__ Vt,
                       int R, int Cc) {
  __shared__ float tile[32][33];   // only shared var => LDS offset 0
  const int c0 = (int)blockIdx.x * 32;
  const int r0 = (int)blockIdx.y * 32;
  const int tx = (int)threadIdx.x;   // 0..31
  const int ty = (int)threadIdx.y;   // 0..7

#if HAVE_TDM
  if (ty == 0) {   // wave 0 issues the tensor DMA (EXEC-independent op)
    const unsigned long long ga =
        (unsigned long long)(uintptr_t)(V + (size_t)r0 * (size_t)Cc + c0);
    // D# group 0: count=1 | lds_addr=0 | global_addr | type=2
    u32x4 g0;
    g0[0] = 1u;
    g0[1] = 0u;
    g0[2] = (unsigned)(ga & 0xFFFFFFFFu);
    g0[3] = (unsigned)((ga >> 32) & 0x01FFFFFFu) | (2u << 30);
    // D# group 1: data_size=4B(code 2), pad_enable, pad_interval=32 DW(code 4),
    // pad_amount=1 DW(code 0); tensor_dim0=Cc, tensor_dim1=R;
    // tile_dim0=32, tile_dim1=32; tensor_dim0_stride=Cc.
    i32x8 g1;
    g1[0] = (int)((2u << 16) | (1u << 20) | (4u << 22));
    g1[1] = (int)(((unsigned)Cc & 0xFFFFu) << 16);
    g1[2] = (int)((((unsigned)Cc >> 16) & 0xFFFFu) |
                  (((unsigned)R & 0xFFFFu) << 16));
    g1[3] = (int)((((unsigned)R >> 16) & 0xFFFFu) | (32u << 16));
    g1[4] = 32;
    g1[5] = (int)(unsigned)Cc;
    g1[6] = 0;
    g1[7] = 0;
    i32x4 z4 = {0, 0, 0, 0};   // groups 2/3 unused (2D tensor)
#if defined(__clang_major__) && (__clang_major__ >= 23)
    i32x8 z8 = {};
    __builtin_amdgcn_tensor_load_to_lds(g0, g1, z4, z4, z8, 0);
#else
    __builtin_amdgcn_tensor_load_to_lds(g0, g1, z4, z4, 0);
#endif
    __builtin_amdgcn_s_wait_tensorcnt(0);
  }
  __syncthreads();
#else
#pragma unroll
  for (int j = 0; j < 32; j += 8)
    tile[ty + j][tx] = V[(size_t)(r0 + ty + j) * (size_t)Cc + (c0 + tx)];
  __syncthreads();
#endif

#pragma unroll
  for (int j = 0; j < 32; j += 8)
    Vt[(size_t)(c0 + ty + j) * (size_t)R + (r0 + tx)] = (__bf16)tile[tx][ty + j];
}

// ---------------------------------------------------------------------------
// In-place stable row softmax: one 256-thread block per row of f[rows, n].
// n % 1024 == 0 (n = 16384). Three float4 sweeps: max, exp+sum, scale.
// ---------------------------------------------------------------------------
__global__ __launch_bounds__(256)
void softmax_rows(float* __restrict__ f, int n) {
  const int row = (int)blockIdx.x;
  float* p = f + (size_t)row * (size_t)n;
  __shared__ float red[256];
  const int t = (int)threadIdx.x;

  float m = -3.402823466e38f;
  for (int i = t * 4; i < n; i += 1024) {
    float4 v = *(const float4*)(p + i);
    m = fmaxf(m, fmaxf(fmaxf(v.x, v.y), fmaxf(v.z, v.w)));
  }
  red[t] = m; __syncthreads();
#pragma unroll
  for (int s = 128; s > 0; s >>= 1) {
    if (t < s) red[t] = fmaxf(red[t], red[t + s]);
    __syncthreads();
  }
  m = red[0]; __syncthreads();

  float sum = 0.0f;
  for (int i = t * 4; i < n; i += 1024) {
    float4 v = *(const float4*)(p + i);
    v.x = __expf(v.x - m); v.y = __expf(v.y - m);
    v.z = __expf(v.z - m); v.w = __expf(v.w - m);
    sum += (v.x + v.y) + (v.z + v.w);
    *(float4*)(p + i) = v;
  }
  red[t] = sum; __syncthreads();
#pragma unroll
  for (int s = 128; s > 0; s >>= 1) {
    if (t < s) red[t] += red[t + s];
    __syncthreads();
  }
  const float inv = 1.0f / red[0];

  for (int i = t * 4; i < n; i += 1024) {
    float4 v = *(const float4*)(p + i);
    v.x *= inv; v.y *= inv; v.z *= inv; v.w *= inv;
    *(float4*)(p + i) = v;
  }
}

// ---------------------------------------------------------------------------
// Orchestration
// ---------------------------------------------------------------------------
extern "C" void kernel_launch(void* const* d_in, const int* in_sizes, int n_in,
                              void* d_out, int out_size, void* d_ws, size_t ws_size,
                              hipStream_t stream) {
  (void)in_sizes; (void)n_in; (void)out_size; (void)ws_size;

  constexpr int Bn = 4096, Dd = 768, Ff = 16384, Mm = 256;

  const float* x    = (const float*)d_in[0];
  const float* mb   = (const float*)d_in[1];
  const float* Wq   = (const float*)d_in[2];
  const float* bq   = (const float*)d_in[3];
  const float* Wk   = (const float*)d_in[4];
  const float* bk   = (const float*)d_in[5];
  const float* Wv   = (const float*)d_in[6];
  const float* bv   = (const float*)d_in[7];
  const float* temp = (const float*)d_in[8];

  // Outputs (return order: x, x_hat, f, V), all fp32, concatenated flat.
  float* out_x    = (float*)d_out;
  float* out_xhat = out_x    + (size_t)Bn * Dd;
  float* out_f    = out_xhat + (size_t)Bn * Dd;
  float* out_V    = out_f    + (size_t)Bn * Ff;

  // Scratch: Q (bf16, B x M), K (bf16, F x M), V^T (bf16, D x F)  = ~34 MB
  char* ws = (char*)d_ws;
  __bf16* Qbf = (__bf16*)(ws);
  __bf16* Kbf = (__bf16*)(ws + 2ull * Bn * Mm);
  __bf16* Vt  = (__bf16*)(ws + 2ull * Bn * Mm + 2ull * Ff * Mm);

  const dim3 wave(32, 1, 1);

  // 0) x passthrough
  hipMemcpyAsync(out_x, x, sizeof(float) * (size_t)Bn * Dd,
                 hipMemcpyDeviceToDevice, stream);

  // 1) Q = x @ Wq^T + bq   -> bf16 scratch
  gemm_abt_wmma<float, float, __bf16>
      <<<dim3(Mm / 64, Bn / 16), wave, 0, stream>>>(
          x, Wq, bq, nullptr, 0, Qbf, Bn, Mm, Dd);

  // 2) K = mb @ Wk^T + bk  -> bf16 scratch
  gemm_abt_wmma<float, float, __bf16>
      <<<dim3(Mm / 64, Ff / 16), wave, 0, stream>>>(
          mb, Wk, bk, nullptr, 0, Kbf, Ff, Mm, Dd);

  // 3) V = mb @ Wv^T + bv  -> fp32 output region
  gemm_abt_wmma<float, float, float>
      <<<dim3(Dd / 64, Ff / 16), wave, 0, stream>>>(
          mb, Wv, bv, nullptr, 0, out_V, Ff, Dd, Dd);

  // 4) V^T (bf16) via TDM-staged LDS transpose
  transpose_to_bf16<<<dim3(Dd / 32, Ff / 32), dim3(32, 8), 0, stream>>>(
      out_V, Vt, Ff, Dd);

  // 5) scores = (Q @ K^T) / temperature  -> f region (raw, pre-softmax)
  gemm_abt_wmma<__bf16, __bf16, float>
      <<<dim3(Ff / 64, Bn / 16), wave, 0, stream>>>(
          Qbf, Kbf, nullptr, temp, /*rcp=*/1, out_f, Bn, Ff, Mm);

  // 6) in-place stable row softmax
  softmax_rows<<<Bn, 256, 0, stream>>>(out_f, Ff);

  // 7) x_hat = f @ V = f @ (V^T)^T
  gemm_abt_wmma<float, __bf16, float>
      <<<dim3(Dd / 64, Bn / 16), wave, 0, stream>>>(
          out_f, Vt, nullptr, nullptr, 0, out_xhat, Bn, Dd, Ff);
}
